// DyHeadBlock_3504693314066
// MI455X (gfx1250) — compile-verified, matrix operand order checked by reading, not run on hardware
//
#include <hip/hip_runtime.h>
#include <hip/hip_bf16.h>

typedef __bf16 v16bf __attribute__((ext_vector_type(16)));
typedef float  v8f   __attribute__((ext_vector_type(8)));

#define ST_SUM  0
#define ST_SSQ  512
#define ST_MU   1024
#define ST_RSTD 1056
#define ST_S    1088
#define ST_SIZE 1152   // floats

struct __align__(16) CtEnt { int idx[4]; float w[4]; };

__device__ __forceinline__ float hsig_(float x){ return fminf(fmaxf(x + 3.f, 0.f), 6.f) * (1.f/6.f); }
__device__ __forceinline__ float sigm_(float x){ return 1.f/(1.f + __expf(-x)); }

// ---------------------------------------------------------------------------
// Pack conv weights [nCo,256,3,3] into WMMA A-fragment layout (bf16):
// wpk[(tile*72 + kt)*32 + lane][16], K = tap*256 + c, kt covers 32 K values.
// 16-bit A layout: lane m rows; elems 0..7 -> K = hs*8+e, elems 8..15 -> K = 16+hs*8+(e-8)
// ---------------------------------------------------------------------------
__global__ void pack_w_kernel(const float* __restrict__ w, __bf16* __restrict__ wpk, int nCo){
    int kt = blockIdx.x, tile = blockIdx.y, lane = threadIdx.x;
    int m = lane & 15, hs = lane >> 4;
    int co = tile*16 + m;
    __bf16* dst = wpk + (((size_t)tile*72 + kt)*32 + lane)*16;
#pragma unroll
    for (int e = 0; e < 16; ++e){
        int klocal = (e < 8) ? (hs*8 + e) : (16 + hs*8 + (e-8));
        int K = kt*32 + klocal;
        int tap = K >> 8, c = K & 255;
        float v = (co < nCo) ? w[((size_t)co*256 + c)*9 + tap] : 0.f;
        dst[e] = (__bf16)v;
    }
}

// ---------------------------------------------------------------------------
// Corner tables
// ---------------------------------------------------------------------------
__global__ void corner_identity_kernel(CtEnt* __restrict__ ct, int Ho, int Hi, int total){
    int id = blockIdx.x*blockDim.x + threadIdx.x;
    if (id >= total) return;
    int N = Ho*Ho;
    int pos = id % N; int tap = (id / N) % 9;
    int oy = pos / Ho, ox = pos % Ho;
    int y = oy - 1 + tap/3, x = ox - 1 + tap%3;
    bool v = (y >= 0 && y < Hi && x >= 0 && x < Hi);
    CtEnt e;
    e.idx[0] = v ? y*Hi + x : 0; e.idx[1] = 0; e.idx[2] = 0; e.idx[3] = 0;
    e.w[0] = v ? 1.f : 0.f; e.w[1] = 0.f; e.w[2] = 0.f; e.w[3] = 0.f;
    ct[id] = e;
}

__global__ void corner_deform_kernel(CtEnt* __restrict__ ct, const float* __restrict__ om,
                                     int Hs, int Ho, int Hi, int stride, int doResize, int total){
    int id = blockIdx.x*blockDim.x + threadIdx.x;
    if (id >= total) return;
    int N = Ho*Ho, Ns = Hs*Hs;
    int pos = id % N; int tap = (id / N) % 9; int b = id / (9*N);
    int oy = pos / Ho, ox = pos % Ho;
    const float* omb = om + (size_t)b*27*Ns;
    float offy, offx, mval;
    if (!doResize){
        int p = oy*Hs + ox;
        offy = omb[(size_t)(2*tap  )*Ns + p];
        offx = omb[(size_t)(2*tap+1)*Ns + p];
        mval = sigm_(omb[(size_t)(18+tap)*Ns + p]);
    } else {
        float sc = (float)(Hs-1) / (float)(Ho-1);
        float ty = oy*sc, tx = ox*sc;
        int y0 = (int)ty, x0 = (int)tx;
        int y1 = min(y0+1, Hs-1), x1 = min(x0+1, Hs-1);
        float wy = ty - y0, wx = tx - x0;
        float w00=(1-wy)*(1-wx), w01=(1-wy)*wx, w10=wy*(1-wx), w11=wy*wx;
        const float* q0 = omb + (size_t)(2*tap)*Ns;
        offy = q0[y0*Hs+x0]*w00 + q0[y0*Hs+x1]*w01 + q0[y1*Hs+x0]*w10 + q0[y1*Hs+x1]*w11;
        const float* q1 = omb + (size_t)(2*tap+1)*Ns;
        offx = q1[y0*Hs+x0]*w00 + q1[y0*Hs+x1]*w01 + q1[y1*Hs+x0]*w10 + q1[y1*Hs+x1]*w11;
        const float* qm = omb + (size_t)(18+tap)*Ns;
        mval = sigm_(qm[y0*Hs+x0])*w00 + sigm_(qm[y0*Hs+x1])*w01
             + sigm_(qm[y1*Hs+x0])*w10 + sigm_(qm[y1*Hs+x1])*w11;
    }
    float py = (float)(oy*stride - 1 + tap/3) + offy;
    float px = (float)(ox*stride - 1 + tap%3) + offx;
    float fy = floorf(py), fx = floorf(px);
    float wy = py - fy, wx = px - fx;
    int iy0 = (int)fy, ix0 = (int)fx;
    float bw[4] = {(1-wy)*(1-wx), (1-wy)*wx, wy*(1-wx), wy*wx};
    CtEnt e;
#pragma unroll
    for (int j = 0; j < 4; ++j){
        int yy = iy0 + (j>>1), xx = ix0 + (j&1);
        bool vld = (yy >= 0 && yy < Hi && xx >= 0 && xx < Hi);
        int cy = min(max(yy,0), Hi-1), cx = min(max(xx,0), Hi-1);
        e.idx[j] = cy*Hi + cx;
        e.w[j]   = vld ? bw[j]*mval : 0.f;
    }
    ct[id] = e;
}

// ---------------------------------------------------------------------------
// Fused deform-sample + WMMA GEMM.
// NW waves x TPW co-tiles per wave (compile-time -> straight-line WMMAs).
// Per K-tile (1 tap x 32 channels): all threads cooperatively gather the
// 32x16 bf16 B-fragment into LDS in WMMA B layout, then each wave issues TPW
// WMMAs reusing it (gathers amortized over NW*TPW output tiles).
// ---------------------------------------------------------------------------
template<int NW, int TPW>
__launch_bounds__(NW*32)
__global__ void deform_gemm_kernel(const float* __restrict__ x,      // [B,256,HiWi]
                                   const CtEnt* __restrict__ ct,     // [B,9,N]
                                   const __bf16* __restrict__ wpk,   // packed A
                                   const float* __restrict__ bias,   // may be null
                                   float* __restrict__ out,          // [B,CoStore,N]
                                   int N, int HiWi, int CoStore){
    constexpr int U = 512/(NW*32);       // B elements per thread
    __shared__ __align__(32) __bf16 lB[512];
    int posBase = blockIdx.x * 16;
    int b = blockIdx.y;
    int tid = threadIdx.x, lane = tid & 31, wave = tid >> 5;
    const float* xb = x + (size_t)b*256*HiWi;

    // this thread's fixed k-local slot for the B build (cl constant per thread)
    int cl = (tid*U) >> 4;
    int hsW, elW;
    if (cl < 16){ hsW = cl >> 3; elW = cl & 7; }
    else        { int c2 = cl - 16; hsW = c2 >> 3; elW = 8 + (c2 & 7); }

    v8f acc[TPW];
#pragma unroll
    for (int a = 0; a < TPW; ++a)
#pragma unroll
        for (int r = 0; r < 8; ++r) acc[a][r] = 0.f;

    for (int kt = 0; kt < 72; ++kt){
        int tap = kt >> 3;
        int cBase = (kt & 7) * 32;
        const CtEnt* ctp = ct + (size_t)(b*9 + tap)*N + posBase;
        const float* xc = xb + (size_t)(cBase + cl)*HiWi;
        // ---- build B fragment (32 K x 16 pos) into LDS ----
#pragma unroll
        for (int u = 0; u < U; ++u){
            int pl = (tid*U + u) & 15;
            const CtEnt e = ctp[pl];
            float v = e.w[0]*xc[e.idx[0]] + e.w[1]*xc[e.idx[1]]
                    + e.w[2]*xc[e.idx[2]] + e.w[3]*xc[e.idx[3]];
            lB[(hsW*16 + pl)*16 + elW] = (__bf16)v;
        }
        __syncthreads();
        v16bf bfrag = *(const v16bf*)&lB[lane*16];
        v16bf afr[TPW];
#pragma unroll
        for (int a = 0; a < TPW; ++a)
            afr[a] = *(const v16bf*)(wpk + (((size_t)(wave*TPW + a)*72 + kt)*32 + lane)*16);
#pragma unroll
        for (int a = 0; a < TPW; ++a)
            acc[a] = __builtin_amdgcn_wmma_f32_16x16x32_bf16(
                false, afr[a], false, bfrag, (short)0, acc[a], false, false);
        __syncthreads();
    }
    // epilogue: C/D layout: elem r -> row m = (lane/16)*8 + r, col n = lane%16
    int n = lane & 15, hsL = lane >> 4;
#pragma unroll
    for (int a = 0; a < TPW; ++a){
        int tile = wave*TPW + a;
#pragma unroll
        for (int r = 0; r < 8; ++r){
            int co = tile*16 + hsL*8 + r;
            if (co < CoStore){
                float v = acc[a][r];
                if (bias) v += bias[co];
                out[((size_t)b*CoStore + co)*N + posBase + n] = v;
            }
        }
    }
}

// ---------------------------------------------------------------------------
// Per-channel sum / sumsq   (block per (b,c), 128 threads)
// ---------------------------------------------------------------------------
__global__ void chan_sums_kernel(const float* __restrict__ f, float* __restrict__ st, int N){
    int bc = blockIdx.x;
    const float* p = f + (size_t)bc*N;
    float s = 0.f, q = 0.f;
    for (int i = threadIdx.x; i < N; i += 128){ float v = p[i]; s += v; q += v*v; }
    __shared__ float rs[128], rq[128];
    rs[threadIdx.x] = s; rq[threadIdx.x] = q; __syncthreads();
    for (int stp = 64; stp > 0; stp >>= 1){
        if (threadIdx.x < stp){ rs[threadIdx.x] += rs[threadIdx.x+stp]; rq[threadIdx.x] += rq[threadIdx.x+stp]; }
        __syncthreads();
    }
    if (threadIdx.x == 0){ st[ST_SUM + bc] = rs[0]; st[ST_SSQ + bc] = rq[0]; }
}

__global__ void gn_stats_kernel(float* __restrict__ st, int N){
    int g = threadIdx.x;
    if (g >= 32) return;
    int b = g >> 4, gr = g & 15;
    float s = 0.f, q = 0.f;
    for (int c = 0; c < 16; ++c){ int bc = b*256 + gr*16 + c; s += st[ST_SUM+bc]; q += st[ST_SSQ+bc]; }
    float cnt = 16.f * (float)N;
    float m = s / cnt;
    float var = q / cnt - m*m;
    st[ST_MU + g] = m;
    st[ST_RSTD + g] = rsqrtf(var + 1e-5f);
}

__global__ void attn_kernel(float* __restrict__ st, const float* __restrict__ gam,
                            const float* __restrict__ bet, const float* __restrict__ wsc,
                            const float* __restrict__ bsc, int N){
    int b = blockIdx.x, c = threadIdx.x;       // 256 threads
    int g = b*16 + (c >> 4);
    float cm = st[ST_SUM + b*256 + c] / (float)N;
    float pn = (cm - st[ST_MU+g]) * st[ST_RSTD+g] * gam[c] + bet[c];
    __shared__ float red[256];
    red[c] = pn * wsc[c]; __syncthreads();
    for (int s = 128; s > 0; s >>= 1){ if (c < s) red[c] += red[c+s]; __syncthreads(); }
    if (c == 0){
        float z = fmaxf(red[0] + bsc[0], 0.f);
        st[ST_S + b] = hsig_(z);
    }
}

// ---------------------------------------------------------------------------
// Combine: inline GN + attn scale for mid/low, inline bilinear upsample + GN
// + attn for hi. outPre = (sum)/n.
// ---------------------------------------------------------------------------
__device__ __forceinline__ float gnorm_(float v, const float* st, int g,
                                        const float* gam, const float* bet, int c){
    return (v - st[ST_MU+g]) * st[ST_RSTD+g] * gam[c] + bet[c];
}

__global__ void combine_kernel(const float* __restrict__ mid, const float* __restrict__ stM,
                               const float* __restrict__ gM,  const float* __restrict__ bM,
                               const float* __restrict__ low, const float* __restrict__ stL,
                               const float* __restrict__ gL,  const float* __restrict__ bL,
                               const float* __restrict__ hi,  const float* __restrict__ stH,
                               const float* __restrict__ gH,  const float* __restrict__ bH, int Ht,
                               float* __restrict__ outPre, int H, int N, float invn){
    int id = blockIdx.x*blockDim.x + threadIdx.x;
    int total = 2*256*N;
    if (id >= total) return;
    int p = id % N; int c = (id / N) & 255; int b = id / (256*N);
    int g = b*16 + (c >> 4);
    float acc = gnorm_(mid[id], stM, g, gM, bM, c) * stM[ST_S + b];
    if (low) acc += gnorm_(low[id], stL, g, gL, bL, c) * stL[ST_S + b];
    if (hi){
        int y = p / H, x = p % H;
        float sc = (float)(Ht-1) / (float)(H-1);
        float ty = y*sc, tx = x*sc;
        int y0 = (int)ty, x0 = (int)tx;
        int y1 = min(y0+1, Ht-1), x1 = min(x0+1, Ht-1);
        float wy = ty - y0, wx = tx - x0;
        const float* hp = hi + ((size_t)b*256 + c)*Ht*Ht;
        float v = hp[y0*Ht+x0]*(1-wy)*(1-wx) + hp[y0*Ht+x1]*(1-wy)*wx
                + hp[y1*Ht+x0]*wy*(1-wx)     + hp[y1*Ht+x1]*wy*wx;
        acc += gnorm_(v, stH, g, gH, bH, c) * stH[ST_S + b];
    }
    outPre[id] = acc * invn;
}

// ---------------------------------------------------------------------------
// DyReLU MLP (per batch): h = relu(pooled @ w1.T + b1); coef = hsig(h@w2.T+b2)-0.5
// ---------------------------------------------------------------------------
__global__ void dyrelu_mlp_kernel(const float* __restrict__ st, int N,
                                  const float* __restrict__ w1, const float* __restrict__ b1,
                                  const float* __restrict__ w2, const float* __restrict__ b2,
                                  float* __restrict__ coefOut){
    int b = blockIdx.x, tid = threadIdx.x;   // 256 threads
    __shared__ float pooled[256];
    __shared__ float h[64];
    pooled[tid] = st[ST_SUM + b*256 + tid] / (float)N;
    __syncthreads();
    if (tid < 64){
        float a = b1[tid];
        for (int c = 0; c < 256; ++c) a += pooled[c] * w1[(size_t)tid*256 + c];
        h[tid] = fmaxf(a, 0.f);
    }
    __syncthreads();
    for (int o = tid; o < 1024; o += 256){
        float a = b2[o];
        for (int k = 0; k < 64; ++k) a += h[k] * w2[(size_t)o*64 + k];
        coefOut[b*1024 + o] = hsig_(a) - 0.5f;
    }
}

__global__ void dyrelu_apply_kernel(const float* __restrict__ pre, const float* __restrict__ coef,
                                    float* __restrict__ out, int N){
    int id = blockIdx.x*blockDim.x + threadIdx.x;
    int total = 2*256*N;
    if (id >= total) return;
    int c = (id / N) & 255; int b = id / (256*N);
    const float* cf = coef + b*1024;
    float a1 = cf[c]*2.f + 1.f, bb1 = cf[256+c];
    float a2 = cf[512+c]*2.f,   bb2 = cf[768+c];
    float v = pre[id];
    out[id] = fmaxf(v*a1 + bb1, v*a2 + bb2);
}

// ---------------------------------------------------------------------------
static inline int cdiv(int a, int b){ return (a + b - 1) / b; }

extern "C" void kernel_launch(void* const* d_in, const int* in_sizes, int n_in,
                              void* d_out, int out_size, void* d_ws, size_t ws_size,
                              hipStream_t stream){
    (void)in_sizes; (void)n_in; (void)out_size; (void)ws_size;
    const float* x[3] = {(const float*)d_in[0], (const float*)d_in[1], (const float*)d_in[2]};
    const float* w_off   = (const float*)d_in[3];
    const float* b_off   = (const float*)d_in[4];
    const float* w_mid   = (const float*)d_in[5];
    const float* g_mid   = (const float*)d_in[6];
    const float* bt_mid  = (const float*)d_in[7];
    const float* w_low   = (const float*)d_in[8];
    const float* g_low   = (const float*)d_in[9];
    const float* bt_low  = (const float*)d_in[10];
    const float* w_high  = (const float*)d_in[11];
    const float* g_high  = (const float*)d_in[12];
    const float* bt_high = (const float*)d_in[13];
    const float* w_scale = (const float*)d_in[14];
    const float* b_scale = (const float*)d_in[15];
    const float* w1 = (const float*)d_in[16];
    const float* b1 = (const float*)d_in[17];
    const float* w2 = (const float*)d_in[18];
    const float* b2 = (const float*)d_in[19];
    float* dout = (float*)d_out;

    const int Hl[3] = {64, 32, 16};
    const int Nl[3] = {4096, 1024, 256};

    // ---- carve workspace ----
    char* base = (char*)d_ws; size_t cur = 0;
    auto A = [&](size_t bytes)->char*{ char* p = base + cur; cur = (cur + bytes + 255) & ~(size_t)255; return p; };

    __bf16* wpkOff  = (__bf16*)A((size_t)2 *72*32*16*2);
    __bf16* wpkMid  = (__bf16*)A((size_t)16*72*32*16*2);
    __bf16* wpkLow  = (__bf16*)A((size_t)16*72*32*16*2);
    __bf16* wpkHigh = (__bf16*)A((size_t)16*72*32*16*2);

    float* om[3];
    for (int l = 0; l < 3; ++l) om[l] = (float*)A((size_t)2*27*Nl[l]*4);

    CtEnt *ctOff[3], *ctMid[3], *ctLow[3], *ctHigh[2];
    for (int l = 0; l < 3; ++l) ctOff[l] = (CtEnt*)A((size_t)2*9*Nl[l]*sizeof(CtEnt));
    for (int l = 0; l < 3; ++l) ctMid[l] = (CtEnt*)A((size_t)2*9*Nl[l]*sizeof(CtEnt));
    ctLow[0] = nullptr;
    for (int l = 1; l < 3; ++l) ctLow[l] = (CtEnt*)A((size_t)2*9*Nl[l]*sizeof(CtEnt));
    for (int l = 0; l < 2; ++l) ctHigh[l] = (CtEnt*)A((size_t)2*9*Nl[l+1]*sizeof(CtEnt));

    float *rMid[3], *rLow[3] = {nullptr,nullptr,nullptr}, *rHigh[2];
    for (int l = 0; l < 3; ++l) rMid[l] = (float*)A((size_t)2*256*Nl[l]*4);
    for (int l = 1; l < 3; ++l) rLow[l] = (float*)A((size_t)2*256*Nl[l]*4);
    for (int l = 0; l < 2; ++l) rHigh[l] = (float*)A((size_t)2*256*Nl[l+1]*4);

    float *stMid[3], *stLow[3] = {nullptr,nullptr,nullptr}, *stHigh[2], *stPre[3];
    for (int l = 0; l < 3; ++l) stMid[l] = (float*)A(ST_SIZE*4);
    for (int l = 1; l < 3; ++l) stLow[l] = (float*)A(ST_SIZE*4);
    for (int l = 0; l < 2; ++l) stHigh[l] = (float*)A(ST_SIZE*4);
    for (int l = 0; l < 3; ++l) stPre[l] = (float*)A(ST_SIZE*4);

    float* outPre[3];
    for (int l = 0; l < 3; ++l) outPre[l] = (float*)A((size_t)2*256*Nl[l]*4);
    float* coefB[3];
    for (int l = 0; l < 3; ++l) coefB[l] = (float*)A((size_t)2*1024*4);

    // ---- 1. pack weights into WMMA A layout ----
    pack_w_kernel<<<dim3(72, 2),  32, 0, stream>>>(w_off,  wpkOff,  27);
    pack_w_kernel<<<dim3(72, 16), 32, 0, stream>>>(w_mid,  wpkMid,  256);
    pack_w_kernel<<<dim3(72, 16), 32, 0, stream>>>(w_low,  wpkLow,  256);
    pack_w_kernel<<<dim3(72, 16), 32, 0, stream>>>(w_high, wpkHigh, 256);

    // ---- 2. offset convs (identity corner tables + WMMA GEMM, 27 outputs) ----
    for (int l = 0; l < 3; ++l){
        int tot = 2*9*Nl[l];
        corner_identity_kernel<<<cdiv(tot,256), 256, 0, stream>>>(ctOff[l], Hl[l], Hl[l], tot);
        deform_gemm_kernel<2,1><<<dim3(Nl[l]/16, 2), 64, 0, stream>>>(
            x[l], ctOff[l], wpkOff, b_off, om[l], Nl[l], Nl[l], 27);
    }

    // ---- 3. deform corner tables ----
    {
        int t0 = 2*9*Nl[0], t1 = 2*9*Nl[1], t2 = 2*9*Nl[2];
        corner_deform_kernel<<<cdiv(t0,256),256,0,stream>>>(ctMid[0],  om[0], 64,64,64,1,0, t0);
        corner_deform_kernel<<<cdiv(t1,256),256,0,stream>>>(ctHigh[0], om[0], 64,32,32,1,1, t1);
        corner_deform_kernel<<<cdiv(t1,256),256,0,stream>>>(ctMid[1],  om[1], 32,32,32,1,0, t1);
        corner_deform_kernel<<<cdiv(t1,256),256,0,stream>>>(ctLow[1],  om[1], 32,32,64,2,0, t1);
        corner_deform_kernel<<<cdiv(t2,256),256,0,stream>>>(ctHigh[1], om[1], 32,16,16,1,1, t2);
        corner_deform_kernel<<<cdiv(t2,256),256,0,stream>>>(ctMid[2],  om[2], 16,16,16,1,0, t2);
        corner_deform_kernel<<<cdiv(t2,256),256,0,stream>>>(ctLow[2],  om[2], 16,16,32,2,0, t2);
    }

    // ---- 4. deform conv GEMMs (fused sampling, WMMA, 16 co-tiles straight-line) ----
    deform_gemm_kernel<4,4><<<dim3(256,2),128,0,stream>>>(x[0], ctMid[0],  wpkMid,  nullptr, rMid[0],  4096, 4096, 256);
    deform_gemm_kernel<4,4><<<dim3(64, 2),128,0,stream>>>(x[1], ctHigh[0], wpkHigh, nullptr, rHigh[0], 1024, 1024, 256);
    deform_gemm_kernel<4,4><<<dim3(64, 2),128,0,stream>>>(x[1], ctMid[1],  wpkMid,  nullptr, rMid[1],  1024, 1024, 256);
    deform_gemm_kernel<4,4><<<dim3(64, 2),128,0,stream>>>(x[0], ctLow[1],  wpkLow,  nullptr, rLow[1],  1024, 4096, 256);
    deform_gemm_kernel<4,4><<<dim3(16, 2),128,0,stream>>>(x[2], ctHigh[1], wpkHigh, nullptr, rHigh[1], 256,  256,  256);
    deform_gemm_kernel<4,4><<<dim3(16, 2),128,0,stream>>>(x[2], ctMid[2],  wpkMid,  nullptr, rMid[2],  256,  256,  256);
    deform_gemm_kernel<4,4><<<dim3(16, 2),128,0,stream>>>(x[1], ctLow[2],  wpkLow,  nullptr, rLow[2],  256,  1024, 256);

    // ---- 5. GN stats + scale-attention scalars for each conv output ----
    auto statsAttn = [&](const float* raw, float* st, int N, const float* gam, const float* bet){
        chan_sums_kernel<<<512, 128, 0, stream>>>(raw, st, N);
        gn_stats_kernel<<<1, 32, 0, stream>>>(st, N);
        attn_kernel<<<2, 256, 0, stream>>>(st, gam, bet, w_scale, b_scale, N);
    };
    statsAttn(rMid[0],  stMid[0],  4096, g_mid,  bt_mid);
    statsAttn(rHigh[0], stHigh[0], 1024, g_high, bt_high);
    statsAttn(rMid[1],  stMid[1],  1024, g_mid,  bt_mid);
    statsAttn(rLow[1],  stLow[1],  1024, g_low,  bt_low);
    statsAttn(rHigh[1], stHigh[1], 256,  g_high, bt_high);
    statsAttn(rMid[2],  stMid[2],  256,  g_mid,  bt_mid);
    statsAttn(rLow[2],  stLow[2],  256,  g_low,  bt_low);

    // ---- 6. combine (inline GN, inline hi-upsample, attn scale, /n) ----
    combine_kernel<<<cdiv(2*256*4096,256),256,0,stream>>>(
        rMid[0], stMid[0], g_mid, bt_mid,
        nullptr, nullptr, nullptr, nullptr,
        rHigh[0], stHigh[0], g_high, bt_high, 32,
        outPre[0], 64, 4096, 0.5f);
    combine_kernel<<<cdiv(2*256*1024,256),256,0,stream>>>(
        rMid[1], stMid[1], g_mid, bt_mid,
        rLow[1], stLow[1], g_low, bt_low,
        rHigh[1], stHigh[1], g_high, bt_high, 16,
        outPre[1], 32, 1024, 1.f/3.f);
    combine_kernel<<<cdiv(2*256*256,256),256,0,stream>>>(
        rMid[2], stMid[2], g_mid, bt_mid,
        rLow[2], stLow[2], g_low, bt_low,
        nullptr, nullptr, nullptr, nullptr, 1,
        outPre[2], 16, 256, 0.5f);

    // ---- 7. DyReLU ----
    size_t outOff[3] = {0, (size_t)2*256*4096, (size_t)2*256*(4096+1024)};
    for (int l = 0; l < 3; ++l){
        chan_sums_kernel<<<512, 128, 0, stream>>>(outPre[l], stPre[l], Nl[l]);
        dyrelu_mlp_kernel<<<2, 256, 0, stream>>>(stPre[l], Nl[l], w1, b1, w2, b2, coefB[l]);
        dyrelu_apply_kernel<<<cdiv(2*256*Nl[l],256), 256, 0, stream>>>(
            outPre[l], coefB[l], dout + outOff[l], Nl[l]);
    }
}
